// MultiheadAttention_2388001816882
// MI455X (gfx1250) — matrix operand order repaired and checked by reading, not compiled, not run on hardware
//
#include <hip/hip_runtime.h>

// ---------------------------------------------------------------------------
// Problem constants (from the reference)
// ---------------------------------------------------------------------------
constexpr int Bc   = 2;
constexpr int Tc   = 2048;
constexpr int Cdim = 1024;
constexpr int Hc   = 16;
constexpr int HDc  = 64;          // head dim
constexpr int Mtot = Bc * Tc;     // 4096 rows for the projection GEMMs

typedef __attribute__((ext_vector_type(4)))  _Float16 hv4;
typedef __attribute__((ext_vector_type(8)))  _Float16 hv8;
typedef __attribute__((ext_vector_type(16))) _Float16 hv16;
typedef __attribute__((ext_vector_type(8)))  float    fv8;

// Matches the builtin's parameter pointee type exactly:
// 'int __attribute__((vector_size(4 * sizeof(int))))'
typedef int v4i_b128 __attribute__((vector_size(16)));

// ---------------------------------------------------------------------------
// CDNA5 async global->LDS copy (ASYNCcnt). Guarded so the file compiles even
// if the toolchain lacks the builtins (falls back to synchronous LDS stores).
// ---------------------------------------------------------------------------
#if defined(__gfx1250__) &&                                                   \
    __has_builtin(__builtin_amdgcn_global_load_async_to_lds_b128) &&          \
    __has_builtin(__builtin_amdgcn_s_wait_asynccnt)
#define USE_ASYNC_COPY 1
#else
#define USE_ASYNC_COPY 0
#endif

__device__ __forceinline__ void lds_copy16_async(_Float16* dst,
                                                 const _Float16* src) {
#if USE_ASYNC_COPY
  // AS1 pointer keeps the 64-bit global address; AS3 pointer is the low
  // 32 bits of the flat address, which is the LDS byte offset (ISA 10.2).
  __builtin_amdgcn_global_load_async_to_lds_b128(
      (__attribute__((address_space(1))) v4i_b128*)(uintptr_t)src,
      (__attribute__((address_space(3))) v4i_b128*)(uintptr_t)dst, 0, 0);
#else
  *reinterpret_cast<hv8*>(dst) = *reinterpret_cast<const hv8*>(src);
#endif
}

__device__ __forceinline__ void wait_async_copies() {
#if USE_ASYNC_COPY
  __builtin_amdgcn_s_wait_asynccnt(0);
#endif
}

// ---------------------------------------------------------------------------
// WMMA helpers (CDNA5: V_WMMA_F32_16X16X32_F16, wave32)
// ---------------------------------------------------------------------------
__device__ __forceinline__ fv8 wmma16(hv16 a, hv16 b, fv8 c) {
  return __builtin_amdgcn_wmma_f32_16x16x32_f16(false, a, false, b, (short)0, c,
                                                false, false);
}

// A-fragment, 16x32 f16 from row-major [M][K] LDS tile -> 2x ds_load_b128.
__device__ __forceinline__ hv16 ld_a_frag(const _Float16* base, int ld,
                                          int m0, int k0) {
  const int lane = threadIdx.x & 31;
  const int rl = lane & 15, hl = lane >> 4;
  const _Float16* row = base + (size_t)(m0 + rl) * ld + k0 + 8 * hl;
  hv8 lo = *reinterpret_cast<const hv8*>(row);        // K = k0+8hl .. +7
  hv8 hi = *reinterpret_cast<const hv8*>(row + 16);   // K = k0+16+8hl .. +7
  return __builtin_shufflevector(lo, hi, 0, 1, 2, 3, 4, 5, 6, 7,
                                 8, 9, 10, 11, 12, 13, 14, 15);
}

// B-fragment, 32x16 f16 (K x N) from TRANSPOSED [N][K] LDS tile:
// per-lane K = k0+8hl..+7 and k0+16+8hl..+7 -> 2x ds_load_b128 (ld%8==0).
__device__ __forceinline__ hv16 ld_bT_frag(const _Float16* baseT, int ld,
                                           int k0, int n0) {
  const int lane = threadIdx.x & 31;
  const int nl = lane & 15, hl = lane >> 4;
  const _Float16* p = baseT + (size_t)(n0 + nl) * ld + k0 + 8 * hl;
  hv8 lo = *reinterpret_cast<const hv8*>(p);
  hv8 hi = *reinterpret_cast<const hv8*>(p + 16);
  return __builtin_shufflevector(lo, hi, 0, 1, 2, 3, 4, 5, 6, 7,
                                 8, 9, 10, 11, 12, 13, 14, 15);
}

// ---------------------------------------------------------------------------
// Kernel 1: convert the four weight matrices f32 -> f16
// ---------------------------------------------------------------------------
__global__ __launch_bounds__(256) void cvt_weights(
    const float* __restrict__ w0, const float* __restrict__ w1,
    const float* __restrict__ w2, const float* __restrict__ w3,
    _Float16* __restrict__ dst) {
  const float* src = (blockIdx.y == 0) ? w0
                   : (blockIdx.y == 1) ? w1
                   : (blockIdx.y == 2) ? w2 : w3;
  _Float16* out = dst + (size_t)blockIdx.y * (Cdim * Cdim);
  size_t i = ((size_t)blockIdx.x * 256 + threadIdx.x) * 4;
  float4 v = *reinterpret_cast<const float4*>(src + i);
  hv4 h = hv4{(_Float16)v.x, (_Float16)v.y, (_Float16)v.z, (_Float16)v.w};
  *reinterpret_cast<hv4*>(out + i) = h;
}

// ---------------------------------------------------------------------------
// Kernel 2/5: tiled WMMA GEMM  out = (A @ W + bias) * scale
//   MODE 0: output f16 in [B, H, T, HD] layout (QKV projection, f32 input)
//   MODE 1: output f32 flat [Mtot, C]          (final projection, f16 input)
// block 256 (8 waves), tile 128x128, K-step 64. B staged transposed.
// ---------------------------------------------------------------------------
constexpr int BM = 128, BN = 128, BKs = 64;
constexpr int LDA_S = BKs + 8;    // 72 halves
constexpr int LDBT  = BKs + 8;    // 72 halves, BsT is [BN][BKs]

template <typename TA, int MODE>
__global__ __launch_bounds__(256) void gemm_wmma_kernel(
    const TA* __restrict__ A, const _Float16* __restrict__ W,
    const float* __restrict__ bias, void* __restrict__ outp, float scale) {
  __shared__ _Float16 As[BM * LDA_S];     // [128][72] row-major [M][K]
  __shared__ _Float16 BsT[BN * LDBT];     // [128][72] transposed [N][K]

  const int tid  = threadIdx.x;
  const int lane = tid & 31;
  const int wave = tid >> 5;        // 0..7
  const int wr   = wave >> 2;       // 0..1 -> 64-row stripe
  const int wc   = wave & 3;        // 0..3 -> 32-col stripe
  const int rl   = lane & 15, hl = lane >> 4;
  const int m_blk = blockIdx.x * BM;
  const int n_blk = blockIdx.y * BN;

  fv8 zero8;
#pragma unroll
  for (int q = 0; q < 8; ++q) zero8[q] = 0.0f;
  fv8 acc[4][2];
#pragma unroll
  for (int i = 0; i < 4; ++i)
#pragma unroll
    for (int j = 0; j < 2; ++j) acc[i][j] = zero8;

  for (int k0 = 0; k0 < Cdim; k0 += BKs) {
    // ---- stage A tile [M][K] ----
    if constexpr (sizeof(TA) == 2) {
      // pure f16 copy: async DMA straight into LDS, 16B per lane
#pragma unroll
      for (int it = 0; it < (BM * BKs) / (256 * 8); ++it) {
        int idx = (it * 256 + tid) * 8;
        int rr = idx >> 6;            // / BKs
        int cc = idx & (BKs - 1);
        lds_copy16_async(
            &As[rr * LDA_S + cc],
            (const _Float16*)A + (size_t)(m_blk + rr) * Cdim + k0 + cc);
      }
    } else {
      // f32 input: convert on the way into LDS
#pragma unroll
      for (int it = 0; it < (BM * BKs) / (256 * 4); ++it) {
        int idx = (it * 256 + tid) * 4;
        int rr = idx >> 6;
        int cc = idx & (BKs - 1);
        const TA* src = A + (size_t)(m_blk + rr) * Cdim + k0 + cc;
        float4 v = *reinterpret_cast<const float4*>(src);
        hv4 h = hv4{(_Float16)v.x, (_Float16)v.y, (_Float16)v.z, (_Float16)v.w};
        *reinterpret_cast<hv4*>(&As[rr * LDA_S + cc]) = h;
      }
    }
    // ---- stage B tile transposed: BsT[n][k] = W[k0+k][n_blk+n] ----
#pragma unroll
    for (int it = 0; it < (BKs * BN) / (256 * 8); ++it) {
      int i = it * 256 + tid;
      int rr = i >> 4;              // k row 0..63
      int cc = (i & 15) * 8;        // n base 0..120
      hv8 w = *reinterpret_cast<const hv8*>(
          &W[(size_t)(k0 + rr) * Cdim + n_blk + cc]);
#pragma unroll
      for (int u = 0; u < 8; ++u) {
        int uu = (u + lane) & 7;    // stagger to spread LDS banks
        BsT[(size_t)(cc + uu) * LDBT + rr] = w[uu];
      }
    }
    if (k0 + BKs < Cdim) {
      __builtin_prefetch(A + (size_t)(m_blk + (tid >> 1)) * Cdim + k0 + BKs +
                             (tid & 1) * 32, 0, 1);
      __builtin_prefetch(W + (size_t)(k0 + BKs + (tid >> 2)) * Cdim + n_blk +
                             (tid & 3) * 32, 0, 1);
    }
    wait_async_copies();
    __syncthreads();

    // ---- 2 WMMA K-steps per staged slab; 16 WMMAs / wave / slab ----
#pragma unroll
    for (int kk = 0; kk < BKs; kk += 32) {
      hv16 af[4], bf[2];
#pragma unroll
      for (int i = 0; i < 4; ++i)
        af[i] = ld_a_frag(As, LDA_S, wr * 64 + i * 16, kk);
#pragma unroll
      for (int j = 0; j < 2; ++j)
        bf[j] = ld_bT_frag(BsT, LDBT, kk, wc * 32 + j * 16);
#pragma unroll
      for (int i = 0; i < 4; ++i)
#pragma unroll
        for (int j = 0; j < 2; ++j)
          acc[i][j] = wmma16(af[i], bf[j], acc[i][j]);
    }
    __syncthreads();
  }

  // ---- epilogue: bias, scale, layout-specific store ----
#pragma unroll
  for (int i = 0; i < 4; ++i) {
#pragma unroll
    for (int j = 0; j < 2; ++j) {
      int n = n_blk + wc * 32 + j * 16 + rl;
      float bval = bias[n];
#pragma unroll
      for (int jr = 0; jr < 8; ++jr) {
        int m = m_blk + wr * 64 + i * 16 + jr + 8 * hl;
        float v = (acc[i][j][jr] + bval) * scale;
        if constexpr (MODE == 0) {
          int b = m >> 11;          // / T (2048)
          int t = m & (Tc - 1);
          int hh = n >> 6;          // / HD (64)
          int d = n & (HDc - 1);
          ((_Float16*)outp)[(((size_t)(b * Hc + hh) * Tc) + t) * HDc + d] =
              (_Float16)v;
        } else {
          ((float*)outp)[(size_t)m * Cdim + n] = v;
        }
      }
    }
  }
}

// ---------------------------------------------------------------------------
// Kernel 3: flash-attention block. One workgroup = one (b,h) x 128 query rows.
// Q,K,V f16 [B*H, T, HD] -> f16 concat-heads [B*T, C].
// grid (16, 32), block 256 (8 waves), ~172 KB dynamic LDS.
// K block is double-buffered and async-DMA'd so the next block's K loads run
// under softmax + P@V of the current block.
// ---------------------------------------------------------------------------
constexpr int QLD = HDc + 8;       // 72: Qs / Ks row stride
constexpr int VLD = 128 + 8;       // 136: Vt row stride (rows = d, cols = key)
constexpr int SLD = 128 + 4;       // 132 (f32)
constexpr int PLD = 128 + 8;       // 136
constexpr size_t SM_ATTN = (size_t)128 * QLD * 2      // Qs
                         + 2 * (size_t)128 * QLD * 2  // Ks[2]
                         + (size_t)HDc * VLD * 2      // Vt
                         + (size_t)128 * SLD * 4      // Ss
                         + (size_t)128 * PLD * 2      // Ps
                         + 3 * 128 * 4;               // m/l/alpha

__device__ __forceinline__ void stage_k_block(_Float16* KsBuf,
                                              const _Float16* Kp, int kb,
                                              int tid) {
  for (int i = tid; i < (128 * HDc) / 8; i += 256) {
    int idx = i * 8;
    int rr = idx >> 6, cc = idx & 63;
    lds_copy16_async(&KsBuf[rr * QLD + cc],
                     &Kp[(size_t)(kb + rr) * HDc + cc]);
  }
}

__global__ __launch_bounds__(256) void attn_kernel(
    const _Float16* __restrict__ Q, const _Float16* __restrict__ K,
    const _Float16* __restrict__ V, _Float16* __restrict__ Ocat) {
  extern __shared__ char smem[];
  _Float16* Qs  = (_Float16*)smem;                    // [128][72]
  _Float16* Ks0 = Qs + 128 * QLD;                     // [128][72] (buffer 0)
  _Float16* Ks1 = Ks0 + 128 * QLD;                    // [128][72] (buffer 1)
  _Float16* Vt  = Ks1 + 128 * QLD;                    // [64][136] transposed V
  float*    Ss  = (float*)(Vt + HDc * VLD);           // [128][132]
  _Float16* Ps  = (_Float16*)(Ss + 128 * SLD);        // [128][136]
  float*    m_s = (float*)(Ps + 128 * PLD);           // [128]
  float*    l_s = m_s + 128;
  float*    a_s = l_s + 128;

  const int tid  = threadIdx.x;
  const int lane = tid & 31;
  const int wave = tid >> 5;        // 0..7 -> 16-row query stripe
  const int rl   = lane & 15, hl = lane >> 4;
  const int bh = blockIdx.y;        // 0..31
  const int b  = bh >> 4;           // / H
  const int h  = bh & (Hc - 1);
  const int t0 = blockIdx.x * 128;

  const _Float16* Qp = Q + (size_t)bh * Tc * HDc;
  const _Float16* Kp = K + (size_t)bh * Tc * HDc;
  const _Float16* Vp = V + (size_t)bh * Tc * HDc;
  _Float16* KsB[2] = {Ks0, Ks1};

  // stage Q block (async) + first K block (async)
  for (int i = tid; i < (128 * HDc) / 8; i += 256) {
    int idx = i * 8;
    int rr = idx >> 6, cc = idx & 63;
    lds_copy16_async(&Qs[rr * QLD + cc],
                     &Qp[(size_t)(t0 + rr) * HDc + cc]);
  }
  stage_k_block(KsB[0], Kp, 0, tid);
  for (int i = tid; i < 128; i += 256) { m_s[i] = -3.0e38f; l_s[i] = 0.0f; }

  fv8 accO[4];
#pragma unroll
  for (int j = 0; j < 4; ++j)
#pragma unroll
    for (int q = 0; q < 8; ++q) accO[j][q] = 0.0f;

  constexpr int NBLK = Tc / 128;    // 16
  for (int ib = 0; ib < NBLK; ++ib) {
    const int kb = ib * 128;
    const _Float16* Ks = KsB[ib & 1];

    // ---- stage V transposed: Vt[d][key] (staggered scatter) ----
    for (int i = tid; i < (HDc / 8) * 128; i += 256) {
      int kk = i & 127;
      int dg = (i >> 7) * 8;
      hv8 vv = *reinterpret_cast<const hv8*>(&Vp[(size_t)(kb + kk) * HDc + dg]);
#pragma unroll
      for (int u = 0; u < 8; ++u) {
        int uu = (u + lane) & 7;
        Vt[(size_t)(dg + uu) * VLD + kk] = vv[uu];
      }
    }
    wait_async_copies();   // our wave's pending async K (and Q) copies done
    __syncthreads();       // ...and everyone else's

    // ---- S = Q @ K^T : each wave does its 16 rows x 128 keys ----
    const int qr = wave * 16;
#pragma unroll
    for (int nt = 0; nt < 8; ++nt) {
      fv8 s;
#pragma unroll
      for (int q = 0; q < 8; ++q) s[q] = 0.0f;
#pragma unroll
      for (int kk = 0; kk < HDc; kk += 32) {
        hv16 af = ld_a_frag(Qs, QLD, qr, kk);
        hv16 bf = ld_bT_frag(Ks, QLD, kk, nt * 16);
        s = wmma16(af, bf, s);
      }
#pragma unroll
      for (int jr = 0; jr < 8; ++jr)
        Ss[(size_t)(qr + jr + 8 * hl) * SLD + nt * 16 + rl] = s[jr];
    }
    __syncthreads();       // Ss ready; all waves done reading both Ks buffers

    // kick off next K block DMA; it overlaps softmax + P@V below
    if (ib + 1 < NBLK) stage_k_block(KsB[(ib + 1) & 1], Kp, kb + 128, tid);

    // ---- online softmax over this 128-key block ----
    {
      int row = wave * 16 + rl;
      const float* srow = &Ss[(size_t)row * SLD + hl * 64];
      float mloc = -3.0e38f;
#pragma unroll 8
      for (int c = 0; c < 64; ++c) mloc = fmaxf(mloc, srow[c]);
      mloc = fmaxf(mloc, __shfl_xor(mloc, 16, 32));   // combine lane halves
      float mold = m_s[row];
      float mnew = fmaxf(mold, mloc);
      float alpha = __expf(mold - mnew);
      float lsum = 0.0f;
      _Float16* prow = &Ps[(size_t)row * PLD + hl * 64];
#pragma unroll 8
      for (int c = 0; c < 64; ++c) {
        float p = __expf(srow[c] - mnew);
        lsum += p;
        prow[c] = (_Float16)p;
      }
      lsum += __shfl_xor(lsum, 16, 32);
      if (hl == 0) {
        l_s[row] = l_s[row] * alpha + lsum;
        m_s[row] = mnew;
        a_s[row] = alpha;
      }
    }
    __syncthreads();

    // ---- rescale running O, then O += P @ V ----
    float alpha_r[8];
#pragma unroll
    for (int jr = 0; jr < 8; ++jr)
      alpha_r[jr] = a_s[wave * 16 + jr + 8 * hl];
#pragma unroll
    for (int j = 0; j < 4; ++j)
#pragma unroll
      for (int jr = 0; jr < 8; ++jr) accO[j][jr] *= alpha_r[jr];

#pragma unroll
    for (int kk = 0; kk < 128; kk += 32) {
      hv16 af = ld_a_frag(Ps, PLD, wave * 16, kk);
#pragma unroll
      for (int j = 0; j < 4; ++j) {
        hv16 bf = ld_bT_frag(Vt, VLD, kk, j * 16);
        accO[j] = wmma16(af, bf, accO[j]);
      }
    }
    __syncthreads();       // before Vt/Ss/Ps are overwritten next iteration
  }

  // ---- finalize: O /= l, store concat-heads f16 [B*T, C] ----
#pragma unroll
  for (int jr = 0; jr < 8; ++jr) {
    int row = wave * 16 + jr + 8 * hl;
    float inv = 1.0f / l_s[row];
    int t = t0 + row;
    size_t base = ((size_t)b * Tc + t) * Cdim + h * HDc;
#pragma unroll
    for (int j = 0; j < 4; ++j)
      Ocat[base + j * 16 + rl] = (_Float16)(accO[j][jr] * inv);
  }
}

// ---------------------------------------------------------------------------
// Launch
// ---------------------------------------------------------------------------
extern "C" void kernel_launch(void* const* d_in, const int* in_sizes, int n_in,
                              void* d_out, int out_size, void* d_ws,
                              size_t ws_size, hipStream_t stream) {
  const float* queries = (const float*)d_in[0];
  const float* keys    = (const float*)d_in[1];
  const float* values  = (const float*)d_in[2];
  const float* Wq = (const float*)d_in[3];  const float* bq = (const float*)d_in[4];
  const float* Wk = (const float*)d_in[5];  const float* bk = (const float*)d_in[6];
  const float* Wv = (const float*)d_in[7];  const float* bv = (const float*)d_in[8];
  const float* Wo = (const float*)d_in[9];  const float* bo = (const float*)d_in[10];

  constexpr size_t CCn = (size_t)Cdim * Cdim;   // 1M elems per weight
  constexpr size_t BTC = (size_t)Mtot * Cdim;   // 4M elems per activation

  _Float16* w16  = (_Float16*)d_ws;
  _Float16* Wq16 = w16;
  _Float16* Wk16 = w16 + CCn;
  _Float16* Wv16 = w16 + 2 * CCn;
  _Float16* Wo16 = w16 + 3 * CCn;
  _Float16* Q16  = w16 + 4 * CCn;
  _Float16* K16  = Q16 + BTC;
  _Float16* V16  = K16 + BTC;
  _Float16* ATT  = V16 + BTC;                    // concat-heads f16 [B*T, C]

  (void)hipFuncSetAttribute((const void*)attn_kernel,
                            hipFuncAttributeMaxDynamicSharedMemorySize,
                            (int)SM_ATTN);

  // 1) weights f32 -> f16
  cvt_weights<<<dim3((unsigned)(CCn / (4 * 256)), 4), 256, 0, stream>>>(
      Wq, Wk, Wv, Wo, w16);

  // 2) Q/K/V projections (scale 1/sqrt(HD) folded into Q)
  dim3 ggrid(Mtot / BM, Cdim / BN);
  gemm_wmma_kernel<float, 0><<<ggrid, 256, 0, stream>>>(
      queries, Wq16, bq, (void*)Q16, 0.125f);
  gemm_wmma_kernel<float, 0><<<ggrid, 256, 0, stream>>>(
      keys, Wk16, bk, (void*)K16, 1.0f);
  gemm_wmma_kernel<float, 0><<<ggrid, 256, 0, stream>>>(
      values, Wv16, bv, (void*)V16, 1.0f);

  // 3) flash attention
  attn_kernel<<<dim3(Tc / 128, Bc * Hc), 256, SM_ATTN, stream>>>(
      Q16, K16, V16, ATT);

  // 4) output projection -> f32 d_out
  gemm_wmma_kernel<_Float16, 1><<<ggrid, 256, 0, stream>>>(
      ATT, Wo16, bo, d_out, 1.0f);
}
